// NamedEntityRecognitionModel_2207613190730
// MI455X (gfx1250) — matrix operand order repaired and checked by reading, not compiled, not run on hardware
//
#include <hip/hip_runtime.h>
#include <hip/hip_bf16.h>

typedef __attribute__((ext_vector_type(16))) _Float16 v16h;
typedef __attribute__((ext_vector_type(8)))  _Float16 v8h;
typedef __attribute__((ext_vector_type(8)))  float    v8f;

#define SQLEN 512
#define BATCH 128
#define DM    512      // d_model == 2*HID
#define HID   256
#define G4    1024     // 4*HID (gate dim per direction)
#define NTAGS 32

// ---------------- WMMA fragment loaders (ISA 7.12.2 layouts) ----------------
// A-matrix 16x32 f16: lane (r = lane&15 is row, h = lane>>4), 8 VGPRs:
// k = 8h + e (e<8), k = 16 + 8h + (e-8) (e>=8)  -> two 16B contiguous loads.
__device__ __forceinline__ v16h load_frag_a(const _Float16* base, int ld, int row0, int k0) {
  const int lane = threadIdx.x & 31;
  const int r = lane & 15, h = lane >> 4;
  const _Float16* p = base + (size_t)(row0 + r) * ld + (k0 + 8 * h);
  v8h lo = *(const v8h*)p;
  v8h hi = *(const v8h*)(p + 16);
  v16h out;
#pragma unroll
  for (int e = 0; e < 8; ++e) { out[e] = lo[e]; out[e + 8] = hi[e]; }
  return out;
}

// B-matrix 32x16 f16 (stored row-major as W[n][k]): lane holds column n=lane&15,
// k = 8h + e (e<8), k = 16 + 8h + (e-8) (e>=8) (two 16B loads).
__device__ __forceinline__ v16h load_frag_b(const _Float16* base, int ld, int col0, int k0) {
  const int lane = threadIdx.x & 31;
  const int c = lane & 15, h = lane >> 4;
  const _Float16* p = base + (size_t)(col0 + c) * ld + (k0 + 8 * h);
  v8h lo = *(const v8h*)p;
  v8h hi = *(const v8h*)(p + 16);
  v16h out;
#pragma unroll
  for (int e = 0; e < 8; ++e) { out[e] = lo[e]; out[e + 8] = hi[e]; }
  return out;
}

// Branch-free fast transcendentals (v_exp_f32 / v_rcp_f32; correct saturation at +-inf).
__device__ __forceinline__ float frcp(float x) { return __builtin_amdgcn_rcpf(x); }
__device__ __forceinline__ float sigf(float x) { return frcp(1.0f + __expf(-x)); }
__device__ __forceinline__ float tanh_fast(float x) {
  return 1.0f - 2.0f * frcp(__expf(2.0f * x) + 1.0f);
}

// ---------------- small prep kernels ----------------
__global__ void f32_to_f16_kernel(const float* __restrict__ in, _Float16* __restrict__ out, size_t n) {
  size_t i = (size_t)blockIdx.x * blockDim.x + threadIdx.x;
  if (i < n) out[i] = (_Float16)in[i];
}

__global__ void bias_combine_kernel(const float* __restrict__ a, const float* __restrict__ b,
                                    float* __restrict__ out, int n) {
  int i = blockIdx.x * blockDim.x + threadIdx.x;
  if (i < n) out[i] = a[i] + b[i];
}

__global__ void embed_kernel(const int* __restrict__ src, const float* __restrict__ emb,
                             _Float16* __restrict__ out) {
  size_t i = (size_t)blockIdx.x * blockDim.x + threadIdx.x;  // over S*B*DM
  int d = (int)(i & (DM - 1));
  size_t sb = i >> 9;                                        // DM == 512
  int tok = src[sb];
  out[i] = (_Float16)(emb[(size_t)tok * DM + d] * 22.627416997969522f);  // sqrt(512)
}

// ---------------- tiled WMMA GEMM: C[m,n] = A[m,:] . Bw[n,:] + bias[n] ----------------
// A: [M][K] f16 row-major, Bw: [N][K] f16 row-major. One wave = 1 m-tile x NT n-tiles.
template <int NT>
__global__ __launch_bounds__(256) void gemm_wmma(
    const _Float16* __restrict__ A, const _Float16* __restrict__ Bw,
    const float* __restrict__ bias,
    _Float16* __restrict__ outH, float* __restrict__ outF,
    int N, int K) {
  const int wave = threadIdx.x >> 5, lane = threadIdx.x & 31;
  const int mt = blockIdx.x * 8 + wave;          // m-tile index
  const int n0 = blockIdx.y * (NT * 16);
  const int cidx = lane & 15, half = lane >> 4;
  const _Float16* Arow = A + (size_t)mt * 16 * K;
  v8f acc[NT] = {};
  for (int k0 = 0; k0 < K; k0 += 32) {
    v16h a = load_frag_a(Arow, K, 0, k0);
#pragma unroll
    for (int nt = 0; nt < NT; ++nt) {
      v16h b = load_frag_b(Bw, K, n0 + nt * 16, k0);
      acc[nt] = __builtin_amdgcn_wmma_f32_16x16x32_f16(false, a, false, b, (short)0,
                                                       acc[nt], false, false);
    }
  }
#pragma unroll
  for (int nt = 0; nt < NT; ++nt) {
    const int n = n0 + nt * 16 + cidx;
    const float bs = bias ? bias[n] : 0.0f;
#pragma unroll
    for (int r = 0; r < 8; ++r) {
      const size_t m = (size_t)mt * 16 + r + 8 * half;   // C layout: row = r + 8*half
      float v = acc[nt][r] + bs;
      if (outH) outH[m * (size_t)N + n] = (_Float16)v;
      else      outF[m * (size_t)N + n] = v;
    }
  }
}

// ---------------- BiLSTM recurrence: one workgroup per direction ----------------
// gates[b,g] = pre[t,b,dir,g] + h_prev[b,:] @ Whh[g,:]^T   (M=128, N=1024, K=256)
// 32 waves: wave = (mt, htg); wave owns all 4 gate tiles for its (mt, ht) so the
// LSTM cell update is register-local; c-state lives in registers for all 512 steps;
// h double-buffered in LDS (2 x 64KB), one barrier per step. Whh fragments are
// re-fetched from WGP$/L2 each step: an opaque SGPR offset (0) is added to the
// weight pointer per step so LICM cannot hoist the invariant loads (which would
// spill 1024 registers to scratch), while keeping global address-space so the
// loads lower to global_load_b128 rather than flat_load.
__global__ __launch_bounds__(1024) void lstm_scan_kernel(
    const _Float16* __restrict__ pre,    // [S*B][2*G4]
    const _Float16* __restrict__ WhhH,   // [2][G4][HID]
    _Float16* __restrict__ xout) {       // [S*B][2*HID]
  const int dir = blockIdx.x;
  const _Float16* W = WhhH + (size_t)dir * G4 * HID;
  extern __shared__ _Float16 hbuf[];     // 2 * BATCH * HID f16 = 128 KB
  const int tid = threadIdx.x;
  for (int i = tid; i < BATCH * HID; i += 1024) hbuf[i] = (_Float16)0.0f;  // h0 = 0
  __syncthreads();

  const int wave = tid >> 5, lane = tid & 31;
  const int mt = wave >> 2;              // 0..7  : rows 16*mt .. 16*mt+15 of batch
  const int htg = wave & 3;              // h-tiles {htg, htg+4, htg+8, htg+12}
  const int cidx = lane & 15, half = lane >> 4;

  float cst[4][8];
#pragma unroll
  for (int a = 0; a < 4; ++a)
#pragma unroll
    for (int r = 0; r < 8; ++r) cst[a][r] = 0.0f;

#pragma unroll 1
  for (int step = 0; step < SQLEN; ++step) {
    const int t = dir ? (SQLEN - 1 - step) : step;
    const _Float16* hprev = hbuf + (size_t)(step & 1) * BATCH * HID;
    _Float16* hnext = hbuf + (size_t)((step + 1) & 1) * BATCH * HID;

    // Opaque zero offset (SGPR): defeats LICM on the step loop while preserving
    // the global address space of W (loads stay global_load_b128 from WGP$/L2).
    unsigned int woff = 0;
    asm volatile("" : "+s"(woff));
    const _Float16* Wstep = W + woff;

#pragma unroll 1
    for (int hi4 = 0; hi4 < 4; ++hi4) {
      const int ht = htg + 4 * hi4;
      v8f acc[4] = {};
#pragma unroll 2
      for (int ks = 0; ks < 8; ++ks) {          // K = 256
        const int k0 = ks * 32;
        v16h a = load_frag_a(hprev, HID, mt * 16, k0);
#pragma unroll
        for (int gate = 0; gate < 4; ++gate) {
          v16h b = load_frag_b(Wstep, HID, gate * HID + ht * 16, k0);
          acc[gate] = __builtin_amdgcn_wmma_f32_16x16x32_f16(false, a, false, b, (short)0,
                                                             acc[gate], false, false);
        }
      }
      const int hcol = ht * 16 + cidx;
#pragma unroll
      for (int r = 0; r < 8; ++r) {
        const int brow = mt * 16 + r + 8 * half;
        const size_t pb = ((size_t)t * BATCH + brow) * (size_t)(2 * G4) + (size_t)dir * G4;
        const float gi = acc[0][r] + (float)pre[pb + 0 * HID + hcol];
        const float gf = acc[1][r] + (float)pre[pb + 1 * HID + hcol];
        const float gg = acc[2][r] + (float)pre[pb + 2 * HID + hcol];
        const float go = acc[3][r] + (float)pre[pb + 3 * HID + hcol];
        float c = sigf(gf) * cst[hi4][r] + sigf(gi) * tanh_fast(gg);
        cst[hi4][r] = c;
        const float h = sigf(go) * tanh_fast(c);
        const _Float16 hh = (_Float16)h;
        hnext[brow * HID + hcol] = hh;
        xout[((size_t)t * BATCH + brow) * (size_t)(2 * HID) + dir * HID + hcol] = hh;
      }
    }
    __syncthreads();   // h writes visible before next step's reads
  }
}

// ---------------- CRF: alpha recursion + gold score, one wave per batch ----------------
__global__ __launch_bounds__(32) void crf_kernel(
    const float* __restrict__ logits,   // [S*B][NTAGS]
    const float* __restrict__ trans,    // [NTAGS][NTAGS], trans[j][k]: j->k
    const int* __restrict__ target,     // [S][B]
    float* __restrict__ perb) {         // [B]
  const int b = blockIdx.x, k = threadIdx.x;
  __shared__ float alpha[NTAGS];
  __shared__ float tr_s[NTAGS * NTAGS];
  for (int i = k; i < NTAGS * NTAGS; i += 32) tr_s[i] = trans[i];
  alpha[k] = logits[b * NTAGS + k];
  __syncthreads();
  for (int s = 1; s < SQLEN; ++s) {
    float m = -3.4e38f;
    for (int j = 0; j < NTAGS; ++j) m = fmaxf(m, alpha[j] + tr_s[j * NTAGS + k]);
    float sum = 0.0f;
    for (int j = 0; j < NTAGS; ++j) sum += __expf(alpha[j] + tr_s[j * NTAGS + k] - m);
    const float na = m + __logf(sum) + logits[((size_t)s * BATCH + b) * NTAGS + k];
    __syncthreads();
    alpha[k] = na;
    __syncthreads();
  }
  // log_z = logsumexp(alpha)
  float a = alpha[k];
  float mm = a;
  for (int off = 16; off; off >>= 1) mm = fmaxf(mm, __shfl_xor(mm, off, 32));
  float e = __expf(a - mm);
  for (int off = 16; off; off >>= 1) e += __shfl_xor(e, off, 32);
  const float logz = mm + __logf(e);
  // gold path score
  float em = 0.0f, tr = 0.0f;
  for (int s = k; s < SQLEN; s += 32) {
    const int tg = target[s * BATCH + b];
    em += logits[((size_t)s * BATCH + b) * NTAGS + tg];
    if (s + 1 < SQLEN) {
      const int tg2 = target[(s + 1) * BATCH + b];
      tr += tr_s[tg * NTAGS + tg2];
    }
  }
  for (int off = 16; off; off >>= 1) { em += __shfl_xor(em, off, 32); tr += __shfl_xor(tr, off, 32); }
  if (k == 0) perb[b] = logz - em - tr;
}

__global__ __launch_bounds__(128) void finalize_kernel(const float* __restrict__ perb,
                                                       float* __restrict__ out) {
  __shared__ float s[128];
  const int t = threadIdx.x;
  s[t] = perb[t];
  __syncthreads();
  for (int off = 64; off; off >>= 1) {
    if (t < off) s[t] += s[t + off];
    __syncthreads();
  }
  if (t == 0) out[0] = s[0] / 128.0f;
}

// ---------------- driver ----------------
extern "C" void kernel_launch(void* const* d_in, const int* in_sizes, int n_in,
                              void* d_out, int out_size, void* d_ws, size_t ws_size,
                              hipStream_t stream) {
  (void)in_sizes; (void)n_in; (void)out_size; (void)ws_size;
  const int*   source = (const int*)d_in[0];
  const int*   target = (const int*)d_in[1];
  const float* emb    = (const float*)d_in[2];
  const float* W_ih   = (const float*)d_in[3];   // [2][2][G4][DM]
  const float* W_hh   = (const float*)d_in[4];   // [2][2][G4][HID]
  const float* b_ih   = (const float*)d_in[5];   // [2][2][G4]
  const float* b_hh   = (const float*)d_in[6];
  const float* W_lin  = (const float*)d_in[7];   // [NTAGS][2*HID]
  const float* b_lin  = (const float*)d_in[8];   // [NTAGS]
  const float* trans  = (const float*)d_in[9];   // [NTAGS][NTAGS]
  float* out = (float*)d_out;

  char* ws = (char*)d_ws;
  size_t off = 0;
  auto alloc = [&](size_t bytes) -> void* {
    void* p = ws + off;
    off += (bytes + 255) & ~(size_t)255;
    return p;
  };
  _Float16* xh0   = (_Float16*)alloc((size_t)SQLEN * BATCH * DM * 2);         // 64 MB
  _Float16* xh1   = (_Float16*)alloc((size_t)SQLEN * BATCH * DM * 2);         // 64 MB
  _Float16* preH  = (_Float16*)alloc((size_t)SQLEN * BATCH * 2 * G4 * 2);     // 256 MB
  _Float16* WihH  = (_Float16*)alloc((size_t)2 * 2 * G4 * DM * 2);            // 4 MB
  _Float16* WhhH  = (_Float16*)alloc((size_t)2 * 2 * G4 * HID * 2);           // 2 MB
  _Float16* WlinH = (_Float16*)alloc((size_t)NTAGS * DM * 2);
  float*    biasF = (float*)alloc((size_t)2 * 2 * G4 * 4);
  float*    logits= (float*)alloc((size_t)SQLEN * BATCH * NTAGS * 4);         // 8 MB
  float*    perb  = (float*)alloc((size_t)BATCH * 4);

  // weight conversions to f16 (layouts are already [n][k] row-major flat)
  { size_t n = (size_t)2 * 2 * G4 * DM;
    f32_to_f16_kernel<<<(unsigned)((n + 255) / 256), 256, 0, stream>>>(W_ih, WihH, n); }
  { size_t n = (size_t)2 * 2 * G4 * HID;
    f32_to_f16_kernel<<<(unsigned)((n + 255) / 256), 256, 0, stream>>>(W_hh, WhhH, n); }
  { size_t n = (size_t)NTAGS * DM;
    f32_to_f16_kernel<<<(unsigned)((n + 255) / 256), 256, 0, stream>>>(W_lin, WlinH, n); }
  { int n = 2 * 2 * G4;
    bias_combine_kernel<<<(n + 255) / 256, 256, 0, stream>>>(b_ih, b_hh, biasF, n); }
  { size_t n = (size_t)SQLEN * BATCH * DM;
    embed_kernel<<<(unsigned)((n + 255) / 256), 256, 0, stream>>>(source, emb, xh0); }

  const int MT = SQLEN * BATCH / 16;            // 4096 m-tiles
  dim3 gpre(MT / 8, (2 * G4) / (8 * 16));       // (512, 16)
  for (int l = 0; l < 2; ++l) {
    const _Float16* xin = l ? xh1 : xh0;
    _Float16* xo        = l ? xh0 : xh1;
    // pre[s,b, dir*G4+g] = x . Wih^T + (b_ih + b_hh)   (M=65536, N=2048, K=512)
    gemm_wmma<8><<<gpre, 256, 0, stream>>>(xin, WihH + (size_t)l * 2 * G4 * DM,
                                           biasF + l * 2 * G4, preH, nullptr,
                                           2 * G4, DM);
    // bidirectional recurrence: block 0 = fwd, block 1 = bwd
    lstm_scan_kernel<<<2, 1024, 2 * BATCH * HID * (int)sizeof(_Float16), stream>>>(
        preH, WhhH + (size_t)l * 2 * G4 * HID, xo);
  }

  // logits = x2 . W_lin^T + b_lin  (M=65536, N=32, K=512), f32 out
  dim3 glog(MT / 8, 1);
  gemm_wmma<2><<<glog, 256, 0, stream>>>(xh0, WlinH, b_lin, nullptr, logits, NTAGS, DM);

  crf_kernel<<<BATCH, 32, 0, stream>>>(logits, trans, target, perb);
  finalize_kernel<<<1, 128, 0, stream>>>(perb, out);
}
